// VehicleModeler_52020643889802
// MI455X (gfx1250) — matrix operand order repaired and checked by reading, not compiled, not run on hardware
//
#include <hip/hip_runtime.h>
#include <hip/hip_bf16.h>

// ---------------- problem constants ----------------
#define B_TOT     2048
#define T_STEPS   128
#define STATE_D   32
#define CMD_D     16
#define IN_D      48      // STATE + CMD
#define HID       1024
#define GATES     4096    // 4 * HID
#define MTILE     32      // batch rows per workgroup
#define NTHREADS  512     // 16 waves * 32 lanes (wave32)
#define NBLOCKS   (B_TOT / MTILE)   // 64

// K layout (combined x||h): chunks of 32
#define KCH_X     2       // 64 padded x columns (48 real + bias col 48 + zeros)
#define KCH_H     32      // 1024 h columns
#define KCH_TOT   34
#define PF_DIST   6       // B-tile prefetch distance (chunks)

// packed weight sizes (bf16 elements)
#define WP   (256 * 34 * 512)   // gates weights: 256 N-tiles x 34 K-chunks x 512
#define WOP  (2 * 32 * 512)     // w_out: 2 N-tiles x 32 K-chunks x 512
#define PACK_TOTAL (WP + WOP)
#define GSTRIDE_B (64 * KCH_TOT * 512 * 2)   // byte distance between gate blocks = 2228224

// LDS layout (bytes) — padded strides for conflict-free ds_load_b128
#define XB_STRIDE 72            // bf16 elems per row (64 used; col48 = 1.0 bias lane)
#define H_STRIDE  1032          // bf16 elems per row (1024 used)
#define C_STRIDE  1032          // f32 elems per row (1024 used)
#define OFF_XBUF  0
#define SZ_XBUF   (MTILE * XB_STRIDE * 2)           // 4608
#define OFF_HA    (OFF_XBUF + SZ_XBUF)              // 4608
#define SZ_H      (MTILE * H_STRIDE * 2)            // 66048
#define OFF_HB    (OFF_HA + SZ_H)                   // 70656
#define OFF_CB    (OFF_HB + SZ_H)                   // 136704
#define SZ_CB     (MTILE * C_STRIDE * 4)            // 132096
#define OFF_ST    (OFF_CB + SZ_CB)                  // 268800
#define SZ_ST     (MTILE * STATE_D * 4)             // 4096
#define OFF_LNG   (OFF_ST + SZ_ST)                  // 272896
#define OFF_LNB   (OFF_LNG + IN_D * 4)              // 273088
#define SMEM_BYTES (OFF_LNB + IN_D * 4)             // 273280  (< 320KB WGP LDS)

typedef __attribute__((ext_vector_type(16))) __bf16 v16bf;
typedef __attribute__((ext_vector_type(8)))  __bf16 v8bf;
typedef __attribute__((ext_vector_type(8)))  float  v8f;

union AFrag { v16bf v; v8bf h[2]; };

static __device__ __forceinline__ float fast_rcp(float x) {
#if __has_builtin(__builtin_amdgcn_rcpf)
  return __builtin_amdgcn_rcpf(x);
#else
  return 1.0f / x;
#endif
}
static __device__ __forceinline__ float fast_sig(float x) {
  return fast_rcp(1.0f + __expf(-x));
}
static __device__ __forceinline__ float fast_tanh(float x) {
#if __has_builtin(__builtin_amdgcn_tanhf)
  return __builtin_amdgcn_tanhf(x);
#else
  return 1.0f - 2.0f * fast_rcp(__expf(2.0f * x) + 1.0f);
#endif
}

static __device__ __forceinline__ v8f wmma_bf16(v16bf a, v16bf b, v8f c) {
  return __builtin_amdgcn_wmma_f32_16x16x32_bf16(
      /*neg_a=*/false, a, /*neg_b=*/false, b,
      /*c_mod=*/(short)0, c, /*reuse_a=*/false, /*reuse_b=*/false);
}

// WGP(CU)-scope prefetch: scope 0 pulls into ALL cache levels (incl. WGP$),
// unlike __builtin_prefetch which lowers to SE/SYS scope (GL2-only, useless
// for L2-resident data). One address covers all 4 gate blocks via literal
// offsets (gate blocks are GSTRIDE_B apart).
static __device__ __forceinline__ void prefetch_gates(const __bf16* p) {
  asm volatile("global_prefetch_b8 %0, off\n\t"
               "global_prefetch_b8 %0, off offset:2228224\n\t"
               "global_prefetch_b8 %0, off offset:4456448\n\t"
               "global_prefetch_b8 %0, off offset:6684672"
               :: "v"(p));
}

// A fragment (16x32 bf16, M x K) from LDS, rows = mf*16 + lane%16.
static __device__ __forceinline__ v16bf load_a_frag(const __bf16* base, int stride,
                                                    int kbase, int mf, int lane) {
  const int row = mf * 16 + (lane & 15);
  const int k   = kbase + ((lane >> 4) << 4);
  const __bf16* p = base + row * stride + k;
  AFrag u;
  u.h[0] = *(const v8bf*)(p);
  u.h[1] = *(const v8bf*)(p + 8);
  return u.v;
}

// B fragment: prepacked lane-major, 512 bf16 per 32x16 tile; lane reads 32B.
static __device__ __forceinline__ v16bf load_b_frag(const __bf16* pack, int tile, int lane) {
  const __bf16* p = pack + (size_t)tile * 512 + lane * 16;
  AFrag u;
  u.h[0] = *(const v8bf*)(p);
  u.h[1] = *(const v8bf*)(p + 8);
  return u.v;
}

// ---------------- packing kernel ----------------
// Combined weight W[4096][1088]: k<48 -> w_ih, k==48 -> b_ih+b_hh (bias column,
// multiplied by constant x[48]=1.0), 48<k<64 -> 0, k>=64 -> w_hh[k-64]
__global__ void vm_pack_kernel(const float* __restrict__ w_ih, const float* __restrict__ w_hh,
                               const float* __restrict__ w_out,
                               const float* __restrict__ b_ih, const float* __restrict__ b_hh,
                               __bf16* __restrict__ wpack, __bf16* __restrict__ wopack) {
  long idx = (long)blockIdx.x * 256 + threadIdx.x;
  if (idx < (long)WP) {
    int tile = (int)(idx >> 9), q = (int)(idx & 511);
    int lane = q >> 4, e = q & 15;
    int n = tile / 34, tk = tile % 34;
    int row = n * 16 + (lane & 15);
    int k = tk * 32 + ((lane >> 4) << 4) + e;
    float v;
    if (k < IN_D)       v = w_ih[row * IN_D + k];
    else if (k == IN_D) v = b_ih[row] + b_hh[row];
    else if (k < 64)    v = 0.0f;
    else                v = w_hh[row * HID + (k - 64)];
    wpack[idx] = (__bf16)v;
  } else if (idx < (long)(WP + WOP)) {
    long p = idx - WP;
    int tile = (int)(p >> 9), q = (int)(p & 511);
    int lane = q >> 4, e = q & 15;
    int n = tile >> 5, tk = tile & 31;
    int row = n * 16 + (lane & 15);
    int k = tk * 32 + ((lane >> 4) << 4) + e;
    wopack[p] = (__bf16)w_out[row * HID + k];
  }
}

// ---------------- main persistent LSTM kernel ----------------
__global__ __launch_bounds__(NTHREADS)
void vm_lstm_kernel(const float* __restrict__ init_state, const float* __restrict__ motor,
                    const float* __restrict__ b_out,
                    const float* __restrict__ ln_g, const float* __restrict__ ln_b,
                    const __bf16* __restrict__ wpack, const __bf16* __restrict__ wopack,
                    float* __restrict__ out) {
  extern __shared__ char smem[];
  __bf16* xbuf = (__bf16*)(smem + OFF_XBUF);
  __bf16* hA   = (__bf16*)(smem + OFF_HA);
  __bf16* hB   = (__bf16*)(smem + OFF_HB);
  float*  cbuf = (float*) (smem + OFF_CB);
  float*  st   = (float*) (smem + OFF_ST);
  float*  lnG  = (float*) (smem + OFF_LNG);
  float*  lnB  = (float*) (smem + OFF_LNB);

  const int tid  = threadIdx.x;
  const int lane = tid & 31;
  const int wave = tid >> 5;
  const int rgbase = blockIdx.x * MTILE;

  // ---- init shared state ----
  for (int i = tid; i < MTILE * STATE_D; i += NTHREADS)
    st[i] = init_state[rgbase * STATE_D + i];
  for (int i = tid; i < MTILE * H_STRIDE; i += NTHREADS) {
    hA[i] = (__bf16)0.0f;
    hB[i] = (__bf16)0.0f;
  }
  for (int i = tid; i < MTILE * C_STRIDE; i += NTHREADS)
    cbuf[i] = 0.0f;
  for (int i = tid; i < MTILE * XB_STRIDE; i += NTHREADS)
    xbuf[i] = (__bf16)0.0f;
  if (tid < IN_D) { lnG[tid] = ln_g[tid]; lnB[tid] = ln_b[tid]; }

  const float bout = (wave < 2) ? b_out[wave * 16 + (lane & 15)] : 0.0f;
  const int mofs = (lane >= 16) ? 8 : 0;   // C-fragment row offset for hi lanes

  __syncthreads();
  // constant bias lane: x[row][48] = 1.0 forever (layernorm writes only cols 0..47)
  if (tid < MTILE) xbuf[tid * XB_STRIDE + IN_D] = (__bf16)1.0f;
  __syncthreads();

  for (int t = 0; t < T_STEPS; ++t) {
    __bf16* hread  = (t & 1) ? hB : hA;
    __bf16* hwrite = (t & 1) ? hA : hB;

    // ---- layernorm of concat(state, cmd) -> xbuf (bf16); one row per lane of wave 0
    if (tid < MTILE) {
      const int row = tid;
      const float* cmdp = motor + ((size_t)(rgbase + row) * T_STEPS + t) * CMD_D;
      float cmdv[CMD_D];
#pragma unroll
      for (int c = 0; c < CMD_D; ++c) cmdv[c] = cmdp[c];
      float s = 0.0f, sq = 0.0f;
#pragma unroll
      for (int c = 0; c < STATE_D; ++c) { float v = st[row * STATE_D + c]; s += v; sq += v * v; }
#pragma unroll
      for (int c = 0; c < CMD_D; ++c)   { float v = cmdv[c]; s += v; sq += v * v; }
      const float m = s * (1.0f / IN_D);
      const float var = sq * (1.0f / IN_D) - m * m;
      const float rstd = rsqrtf(var + 1e-5f);
#pragma unroll
      for (int c = 0; c < STATE_D; ++c) {
        float v = st[row * STATE_D + c];
        xbuf[row * XB_STRIDE + c] = (__bf16)((v - m) * rstd * lnG[c] + lnB[c]);
      }
#pragma unroll
      for (int c = 0; c < CMD_D; ++c) {
        float v = cmdv[c];
        xbuf[row * XB_STRIDE + STATE_D + c] =
            (__bf16)((v - m) * rstd * lnG[STATE_D + c] + lnB[STATE_D + c]);
      }
    }
    __syncthreads();

    // ---- gates: wave owns j-tiles jj = wave*4 + jt; i,f,g,o accumulated together
    for (int jt = 0; jt < 4; ++jt) {
      const int jj = wave * 4 + jt;
      const int jcol = jj * 16 + (lane & 15);
      int tb[4];
#pragma unroll
      for (int g = 0; g < 4; ++g) tb[g] = (g * 64 + jj) * KCH_TOT + KCH_X;

      v8f acc[4][2];
#pragma unroll
      for (int g = 0; g < 4; ++g)
#pragma unroll
        for (int mf = 0; mf < 2; ++mf)
#pragma unroll
          for (int r = 0; r < 8; ++r) acc[g][mf][r] = 0.0f;

      // x part (K chunks 0..1; includes bias via col 48)
#pragma unroll
      for (int tk = 0; tk < KCH_X; ++tk) {
        v16bf a0 = load_a_frag(xbuf, XB_STRIDE, tk * 32, 0, lane);
        v16bf a1 = load_a_frag(xbuf, XB_STRIDE, tk * 32, 1, lane);
#pragma unroll
        for (int g = 0; g < 4; ++g) {
          v16bf b = load_b_frag(wpack, tb[g] - KCH_X + tk, lane);
          acc[g][0] = wmma_bf16(a0, b, acc[g][0]);
          acc[g][1] = wmma_bf16(a1, b, acc[g][1]);
        }
      }

      // h part (32 chunks): explicit modulo-2 software pipeline.
      // B (global, L2) is double-buffered; A (LDS, short latency) is just-in-time.
      v16bf b0[4], b1[4];
#pragma unroll
      for (int g = 0; g < 4; ++g) b0[g] = load_b_frag(wpack, tb[g] + 0, lane);
#pragma unroll
      for (int g = 0; g < 4; ++g) b1[g] = load_b_frag(wpack, tb[g] + 1, lane);

#pragma unroll 1
      for (int hk = 0; hk < KCH_H - 2; hk += 2) {
        // warm WGP$ PF_DIST chunks ahead (one line per lane covers each 1KB tile)
        prefetch_gates(wpack + (size_t)(tb[0] + hk + PF_DIST) * 512 + lane * 16);

        // ---- chunk hk (parity 0): refill loads first, then compute
        v16bf t0[4];
#pragma unroll
        for (int g = 0; g < 4; ++g) t0[g] = load_b_frag(wpack, tb[g] + hk + 2, lane);
        {
          v16bf a0 = load_a_frag(hread, H_STRIDE, hk * 32, 0, lane);
          v16bf a1 = load_a_frag(hread, H_STRIDE, hk * 32, 1, lane);
#pragma unroll
          for (int g = 0; g < 4; ++g) {
            acc[g][0] = wmma_bf16(a0, b0[g], acc[g][0]);
            acc[g][1] = wmma_bf16(a1, b0[g], acc[g][1]);
          }
        }
#pragma unroll
        for (int g = 0; g < 4; ++g) b0[g] = t0[g];

        // ---- chunk hk+1 (parity 1)
        v16bf t1[4];
#pragma unroll
        for (int g = 0; g < 4; ++g) t1[g] = load_b_frag(wpack, tb[g] + hk + 3, lane);
        {
          v16bf a0 = load_a_frag(hread, H_STRIDE, (hk + 1) * 32, 0, lane);
          v16bf a1 = load_a_frag(hread, H_STRIDE, (hk + 1) * 32, 1, lane);
#pragma unroll
          for (int g = 0; g < 4; ++g) {
            acc[g][0] = wmma_bf16(a0, b1[g], acc[g][0]);
            acc[g][1] = wmma_bf16(a1, b1[g], acc[g][1]);
          }
        }
#pragma unroll
        for (int g = 0; g < 4; ++g) b1[g] = t1[g];
      }
      // epilogue: chunks 30 (parity 0) and 31 (parity 1)
      {
        v16bf a0 = load_a_frag(hread, H_STRIDE, (KCH_H - 2) * 32, 0, lane);
        v16bf a1 = load_a_frag(hread, H_STRIDE, (KCH_H - 2) * 32, 1, lane);
#pragma unroll
        for (int g = 0; g < 4; ++g) {
          acc[g][0] = wmma_bf16(a0, b0[g], acc[g][0]);
          acc[g][1] = wmma_bf16(a1, b0[g], acc[g][1]);
        }
      }
      {
        v16bf a0 = load_a_frag(hread, H_STRIDE, (KCH_H - 1) * 32, 0, lane);
        v16bf a1 = load_a_frag(hread, H_STRIDE, (KCH_H - 1) * 32, 1, lane);
#pragma unroll
        for (int g = 0; g < 4; ++g) {
          acc[g][0] = wmma_bf16(a0, b1[g], acc[g][0]);
          acc[g][1] = wmma_bf16(a1, b1[g], acc[g][1]);
        }
      }

      // ---- LSTM nonlinearity; c lives in LDS, matching element across the 4 gate frags
#pragma unroll
      for (int mf = 0; mf < 2; ++mf) {
#pragma unroll
        for (int r = 0; r < 8; ++r) {
          const float iv = acc[0][mf][r];
          const float fv = acc[1][mf][r];
          const float gv = acc[2][mf][r];
          const float ov = acc[3][mf][r];
          const int row = mf * 16 + r + mofs;
          float* cp = &cbuf[row * C_STRIDE + jcol];
          float c = fast_sig(fv) * (*cp) + fast_sig(iv) * fast_tanh(gv);
          *cp = c;
          const float h = fast_sig(ov) * fast_tanh(c);
          hwrite[row * H_STRIDE + jcol] = (__bf16)h;
        }
      }
    }
    __syncthreads();

    // ---- output projection: new_state = state + h @ w_out.T + b_out (waves 0,1)
    if (wave < 2) {
      v8f oa[2];
#pragma unroll
      for (int mf = 0; mf < 2; ++mf)
#pragma unroll
        for (int r = 0; r < 8; ++r) oa[mf][r] = 0.0f;

      v16bf ob0 = load_b_frag(wopack, wave * KCH_H + 0, lane);
      v16bf ob1 = load_b_frag(wopack, wave * KCH_H + 1, lane);
#pragma unroll 1
      for (int hk = 0; hk < KCH_H - 2; hk += 2) {
        v16bf t0 = load_b_frag(wopack, wave * KCH_H + hk + 2, lane);
        {
          v16bf a0 = load_a_frag(hwrite, H_STRIDE, hk * 32, 0, lane);
          v16bf a1 = load_a_frag(hwrite, H_STRIDE, hk * 32, 1, lane);
          oa[0] = wmma_bf16(a0, ob0, oa[0]);
          oa[1] = wmma_bf16(a1, ob0, oa[1]);
        }
        ob0 = t0;
        v16bf t1 = load_b_frag(wopack, wave * KCH_H + hk + 3, lane);
        {
          v16bf a0 = load_a_frag(hwrite, H_STRIDE, (hk + 1) * 32, 0, lane);
          v16bf a1 = load_a_frag(hwrite, H_STRIDE, (hk + 1) * 32, 1, lane);
          oa[0] = wmma_bf16(a0, ob1, oa[0]);
          oa[1] = wmma_bf16(a1, ob1, oa[1]);
        }
        ob1 = t1;
      }
      {
        v16bf a0 = load_a_frag(hwrite, H_STRIDE, (KCH_H - 2) * 32, 0, lane);
        v16bf a1 = load_a_frag(hwrite, H_STRIDE, (KCH_H - 2) * 32, 1, lane);
        oa[0] = wmma_bf16(a0, ob0, oa[0]);
        oa[1] = wmma_bf16(a1, ob0, oa[1]);
      }
      {
        v16bf a0 = load_a_frag(hwrite, H_STRIDE, (KCH_H - 1) * 32, 0, lane);
        v16bf a1 = load_a_frag(hwrite, H_STRIDE, (KCH_H - 1) * 32, 1, lane);
        oa[0] = wmma_bf16(a0, ob1, oa[0]);
        oa[1] = wmma_bf16(a1, ob1, oa[1]);
      }

      const int s = wave * 16 + (lane & 15);
#pragma unroll
      for (int mf = 0; mf < 2; ++mf) {
#pragma unroll
        for (int r = 0; r < 8; ++r) {
          const int row = mf * 16 + r + mofs;
          const float v = st[row * STATE_D + s] + oa[mf][r] + bout;
          st[row * STATE_D + s] = v;
          out[((size_t)(rgbase + row) * T_STEPS + t) * STATE_D + s] = v;
        }
      }
    }
    __syncthreads();
  }
}

extern "C" void kernel_launch(void* const* d_in, const int* in_sizes, int n_in,
                              void* d_out, int out_size, void* d_ws, size_t ws_size,
                              hipStream_t stream) {
  const float* init_state = (const float*)d_in[0];
  const float* motor      = (const float*)d_in[1];
  const float* w_ih       = (const float*)d_in[2];
  const float* w_hh       = (const float*)d_in[3];
  const float* b_ih       = (const float*)d_in[4];
  const float* b_hh       = (const float*)d_in[5];
  const float* w_out      = (const float*)d_in[6];
  const float* b_out      = (const float*)d_in[7];
  const float* ln_g       = (const float*)d_in[8];
  const float* ln_b       = (const float*)d_in[9];
  float* out = (float*)d_out;

  __bf16* wpack  = (__bf16*)d_ws;
  __bf16* wopack = wpack + WP;

  vm_pack_kernel<<<(PACK_TOTAL + 255) / 256, 256, 0, stream>>>(
      w_ih, w_hh, w_out, b_ih, b_hh, wpack, wopack);

  vm_lstm_kernel<<<NBLOCKS, NTHREADS, SMEM_BYTES, stream>>>(
      init_state, motor, b_out, ln_g, ln_b, wpack, wopack, out);
}
